// HypGRU_39934605918678
// MI455X (gfx1250) — compile-verified, matrix operand order checked by reading, not run on hardware
//
#include <hip/hip_runtime.h>
#include <math.h>

// ---------------------------------------------------------------------------
// HypGRU forward for MI455X (gfx1250, wave32, WMMA).
// 6 big GEMMs run through v_wmma_f32_16x16x32_f16 (f16 in, f32 accum).
// GEMM: 128x256 block tile, 8 waves, 64x64 per wave (16 WMMA / K-step),
// global->LDS staging via GLOBAL_LOAD_ASYNC_TO_LDS_B128 when available.
// Elementwise hyperbolic math in fp32, one block per row with LDS reductions.
// ---------------------------------------------------------------------------

typedef __attribute__((ext_vector_type(16))) _Float16 v16h;
typedef __attribute__((ext_vector_type(8)))  _Float16 h8;
typedef __attribute__((ext_vector_type(8)))  float    v8f;
typedef __attribute__((ext_vector_type(4)))  int      v4i;

#define TPB 256
#define GMAX 8   // max elements per thread in row kernels (H/TPB <= 8)

#if __has_builtin(__builtin_amdgcn_global_load_async_to_lds_b128) && \
    __has_builtin(__builtin_amdgcn_s_wait_asynccnt)
#define USE_ASYNC_LDS 1
#else
#define USE_ASYNC_LDS 0
#endif

#if USE_ASYNC_LDS
// ROCm signature (from hipcc diagnostic): param 0 is AS(1) int4*  (global src)
typedef __attribute__((address_space(1))) v4i* g4_ptr;
typedef __attribute__((address_space(3))) v4i* l4_ptr;
__device__ __forceinline__ void async_copy_b128(const void* g, const void* l) {
  // generic LDS pointer low 32 bits == LDS byte offset (aperture rule)
  __builtin_amdgcn_global_load_async_to_lds_b128(
      (g4_ptr)(uintptr_t)g,
      (l4_ptr)(unsigned int)(uintptr_t)l, 0, 0);
}
#endif

// ---------------- helpers ----------------

__device__ __forceinline__ float atanh_c(float x) {
  // reference: arctanh(clip(x, -1+1e-5, 1-1e-5))
  x = fminf(fmaxf(x, -1.0f + 1e-5f), 1.0f - 1e-5f);
  return atanhf(x);
}

__device__ __forceinline__ float rowsum(float v, float* sbuf) {
  const int t = threadIdx.x;
  sbuf[t] = v;
  __syncthreads();
  #pragma unroll
  for (int s = TPB / 2; s >= 1; s >>= 1) {
    if (t < s) sbuf[t] += sbuf[t + s];
    __syncthreads();
  }
  float r = sbuf[0];
  __syncthreads();
  return r;
}

// ---------------- preprocess: exp0(inputs), project(state) ----------------

__global__ __launch_bounds__(TPB) void k_pre(
    const float* __restrict__ inputs, const float* __restrict__ state,
    int D, int H,
    _Float16* __restrict__ Xh, _Float16* __restrict__ Sph,
    float* __restrict__ xnorm, float* __restrict__ snorm,
    float* __restrict__ sscale)
{
  __shared__ float sbuf[TPB];
  const int row = blockIdx.x;
  const int t = threadIdx.x;
  const int pd = D / TPB, ph = H / TPB;
  const float* xr = inputs + (size_t)row * D;
  const float* sr = state  + (size_t)row * H;
  float xv[GMAX], sv[GMAX];
  float sx = 0.f, ss = 0.f;
  for (int j = 0; j < pd; ++j) { float v = xr[t + j*TPB]; xv[j] = v; sx += v*v; }
  for (int j = 0; j < ph; ++j) { float v = sr[t + j*TPB]; sv[j] = v; ss += v*v; }
  sx = rowsum(sx, sbuf);
  ss = rowsum(ss, sbuf);
  // hyp_x = exp0(inputs): tanh(n)/n * v ; norm(hyp_x) = tanh(n)
  float nx  = sqrtf(sx);
  float nx_ = fmaxf(nx, 1e-10f);
  float tnx = tanhf(nx_);
  float ax  = tnx / nx_;
  for (int j = 0; j < pd; ++j)
    Xh[(size_t)row*D + t + j*TPB] = (_Float16)(ax * xv[j]);
  // state_p = project(state, eps=1e-4) = sc * state (scalar sc)
  float ns = sqrtf(ss);
  float mxn = 1.0f - 1e-4f;
  float sc = (ns > mxn) ? (mxn / (ns + 1e-10f)) : 1.0f;
  for (int j = 0; j < ph; ++j)
    Sph[(size_t)row*H + t + j*TPB] = (_Float16)(sc * sv[j]);
  if (t == 0) {
    xnorm[row]  = fmaxf(tnx, 1e-10f);
    snorm[row]  = fmaxf(ns * sc, 1e-10f);
    sscale[row] = sc;
  }
}

// ---------------- weight transpose + f32 -> f16 : W[K][N] -> Wt[N][K] ------

__global__ __launch_bounds__(256) void k_wt(
    const float* __restrict__ W, _Float16* __restrict__ Wt, int K, int N)
{
  __shared__ float tile[32][33];
  const int n0 = blockIdx.x * 32;
  const int k0 = blockIdx.y * 32;
  const int tx = threadIdx.x, ty = threadIdx.y;
  for (int j = ty; j < 32; j += 8)
    tile[j][tx] = W[(size_t)(k0 + j) * N + n0 + tx];
  __syncthreads();
  for (int j = ty; j < 32; j += 8)
    Wt[(size_t)(n0 + j) * K + k0 + tx] = (_Float16)tile[tx][j];
}

// ---------------- WMMA GEMM: C[M][N] = A[M][K] * Bt[N][K]^T ----------------
// A, Bt f16 row-major (K contiguous); C fp32.
// Block: 256 threads = 8 waves; tile 128(M) x 256(N); BK=32.
// Wave grid 2(M) x 4(N); each wave: 4x4 tiles of 16x16 => 16 v8f accumulators.

#define BM 128
#define BN 256
#define BK 32
#define LDS_K 48  // padded f16 row stride: keeps 16B alignment, spreads banks

__global__ __launch_bounds__(TPB) void k_gemm(
    const _Float16* __restrict__ A, const _Float16* __restrict__ Bt,
    float* __restrict__ C, int M, int N, int K)
{
  __shared__ _Float16 sA[BM * LDS_K];
  __shared__ _Float16 sB[BN * LDS_K];
  const int tid  = threadIdx.x;
  const int lane = tid & 31;
  const int wave = tid >> 5;
  const int wm = wave & 1;   // 0..1 -> 64 rows each
  const int wn = wave >> 1;  // 0..3 -> 64 cols each
  const int m0 = blockIdx.y * BM;
  const int n0 = blockIdx.x * BN;

  const v8f zero8 = {0.f,0.f,0.f,0.f,0.f,0.f,0.f,0.f};
  v8f acc[4][4];
  #pragma unroll
  for (int i = 0; i < 4; ++i)
    #pragma unroll
    for (int j = 0; j < 4; ++j) acc[i][j] = zero8;

  const int lrow  = lane & 15;
  const int khalf = (lane < 16) ? 0 : 8;

  for (int k = 0; k < K; k += BK) {
    // stage A: 128x32 f16 (512 x 16B chunks), B: 256x32 f16 (1024 chunks)
#if USE_ASYNC_LDS
    #pragma unroll
    for (int c2 = 0; c2 < 2; ++c2) {
      int c  = tid + c2 * TPB;
      int rr = c >> 2;
      int qq = (c & 3) * 8;
      async_copy_b128(A + (size_t)(m0 + rr) * K + k + qq,
                      sA + rr * LDS_K + qq);
    }
    #pragma unroll
    for (int c2 = 0; c2 < 4; ++c2) {
      int c  = tid + c2 * TPB;
      int rr = c >> 2;
      int qq = (c & 3) * 8;
      async_copy_b128(Bt + (size_t)(n0 + rr) * K + k + qq,
                      sB + rr * LDS_K + qq);
    }
    __builtin_amdgcn_s_wait_asynccnt(0);  // own loads done, then barrier
#else
    #pragma unroll
    for (int c2 = 0; c2 < 2; ++c2) {
      int c  = tid + c2 * TPB;
      int rr = c >> 2;
      int qq = (c & 3) * 8;
      *(h8*)(sA + rr * LDS_K + qq) =
          *(const h8*)(A + (size_t)(m0 + rr) * K + k + qq);
    }
    #pragma unroll
    for (int c2 = 0; c2 < 4; ++c2) {
      int c  = tid + c2 * TPB;
      int rr = c >> 2;
      int qq = (c & 3) * 8;
      *(h8*)(sB + rr * LDS_K + qq) =
          *(const h8*)(Bt + (size_t)(n0 + rr) * K + k + qq);
    }
#endif
    __syncthreads();

    // A fragments: lane<16 -> K 0..7 & 16..23 ; lane>=16 -> K 8..15 & 24..31
    v16h af[4], bf[4];
    #pragma unroll
    for (int i = 0; i < 4; ++i) {
      const _Float16* p = sA + (wm * 64 + i * 16 + lrow) * LDS_K + khalf;
      h8 lo = *(const h8*)p;
      h8 hi = *(const h8*)(p + 16);
      #pragma unroll
      for (int e = 0; e < 8; ++e) { af[i][e] = lo[e]; af[i][8 + e] = hi[e]; }
    }
    #pragma unroll
    for (int j = 0; j < 4; ++j) {
      const _Float16* p = sB + (wn * 64 + j * 16 + lrow) * LDS_K + khalf;
      h8 lo = *(const h8*)p;
      h8 hi = *(const h8*)(p + 16);
      #pragma unroll
      for (int e = 0; e < 8; ++e) { bf[j][e] = lo[e]; bf[j][8 + e] = hi[e]; }
    }

    #pragma unroll
    for (int i = 0; i < 4; ++i)
      #pragma unroll
      for (int j = 0; j < 4; ++j)
        acc[i][j] = __builtin_amdgcn_wmma_f32_16x16x32_f16(
            false, af[i], false, bf[j], (short)0, acc[i][j], false, false);
    __syncthreads();
  }

  // C/D layout: VGPR v -> lanes 0-15: M=v, lanes 16-31: M=v+8; N = lane&15
  #pragma unroll
  for (int i = 0; i < 4; ++i)
    #pragma unroll
    for (int j = 0; j < 4; ++j)
      #pragma unroll
      for (int v = 0; v < 8; ++v) {
        int rm = m0 + wm * 64 + i * 16 + ((lane < 16) ? v : v + 8);
        int cn = n0 + wn * 64 + j * 16 + (lane & 15);
        C[(size_t)rm * N + cn] = acc[i][j][v];
      }
}

// ---------------- gate = sigmoid(log0(project(madd(madd(Th,Tx),b)))) -------

__device__ void gate_eval(
    const float* __restrict__ Gh, const float* __restrict__ Gx,
    const float* __restrict__ b, int per,
    float hn, float xn, float* sbuf, float* g)
{
  const int t = threadIdx.x;
  float u[GMAX], v[GMAX], bv[GMAX];
  float s1 = 0.f, s2 = 0.f;
  for (int j = 0; j < per; ++j) {
    float a = Gh[t + j*TPB], c = Gx[t + j*TPB];
    u[j] = a; v[j] = c; bv[j] = b[t + j*TPB];
    s1 += a*a; s2 += c*c;
  }
  s1 = rowsum(s1, sbuf);
  s2 = rowsum(s2, sbuf);
  // mob_matmul scalings (result norms are scalars: |tanh(...)|)
  float nh = fmaxf(sqrtf(s1), 1e-10f);
  float th = tanhf(nh / hn * atanh_c(hn));
  float ah = th / nh;
  float nx = fmaxf(sqrtf(s2), 1e-10f);
  float tx = tanhf(nx / xn * atanh_c(xn));
  float ax = tx / nx;
  // a1 = mob_add(Th, Tx)
  float uv = 0.f;
  for (int j = 0; j < per; ++j) { u[j] *= ah; v[j] *= ax; uv += u[j]*v[j]; }
  uv = rowsum(uv, sbuf);
  float u2 = th*th, v2 = tx*tx;
  float den = 1.f + 2.f*uv + u2*v2 + 1e-10f;
  float cu = 1.f + 2.f*uv + v2, cv = 1.f - u2;
  float duv = 0.f, d2 = 0.f, db = 0.f;
  for (int j = 0; j < per; ++j) {
    float a1 = (cu*u[j] + cv*v[j]) / den;
    u[j] = a1;
    duv += a1*bv[j]; d2 += a1*a1; db += bv[j]*bv[j];
  }
  duv = rowsum(duv, sbuf);
  d2  = rowsum(d2,  sbuf);
  db  = rowsum(db,  sbuf);
  // a2 = mob_add(a1, b)
  float den2 = 1.f + 2.f*duv + d2*db + 1e-10f;
  float cu2 = 1.f + 2.f*duv + db, cv2 = 1.f - d2;
  float nn = 0.f;
  for (int j = 0; j < per; ++j) {
    float a2 = (cu2*u[j] + cv2*bv[j]) / den2;
    u[j] = a2; nn += a2*a2;
  }
  nn = rowsum(nn, sbuf);
  // project(.., 1e-5) then log0, sigmoid
  float n2 = sqrtf(nn);
  float mb = 1.0f - 1e-5f;
  float ps = (n2 > mb) ? (mb / (n2 + 1e-10f)) : 1.0f;
  float npn = fmaxf(n2 * ps, 1e-10f);
  float lg = atanh_c(npn) / npn;
  for (int j = 0; j < per; ++j) {
    float pre = lg * ps * u[j];
    g[j] = 1.f / (1.f + expf(-pre));
  }
}

__global__ __launch_bounds__(TPB) void k_zgate(
    const float* __restrict__ Gh, const float* __restrict__ Gx,
    const float* __restrict__ b, const float* __restrict__ snorm,
    const float* __restrict__ xnorm, int H, _Float16* __restrict__ zout)
{
  __shared__ float sbuf[TPB];
  const int row = blockIdx.x;
  const int per = H / TPB;
  float g[GMAX];
  gate_eval(Gh + (size_t)row*H, Gx + (size_t)row*H, b, per,
            snorm[row], xnorm[row], sbuf, g);
  for (int j = 0; j < per; ++j)
    zout[(size_t)row*H + threadIdx.x + j*TPB] = (_Float16)g[j];
}

__global__ __launch_bounds__(TPB) void k_rgate(
    const float* __restrict__ Gh, const float* __restrict__ Gx,
    const float* __restrict__ b, const float* __restrict__ state,
    const float* __restrict__ sscale, const float* __restrict__ snorm,
    const float* __restrict__ xnorm, int H,
    _Float16* __restrict__ rp, float* __restrict__ rpnorm)
{
  __shared__ float sbuf[TPB];
  const int row = blockIdx.x;
  const int t = threadIdx.x;
  const int per = H / TPB;
  float g[GMAX];
  gate_eval(Gh + (size_t)row*H, Gx + (size_t)row*H, b, per,
            snorm[row], xnorm[row], sbuf, g);
  // r_point_h = project(mob_pw(state_p, r), 1e-4)
  float sc = sscale[row];
  float hn = snorm[row];
  float mxv[GMAX];
  float smm = 0.f;
  for (int j = 0; j < per; ++j) {
    float sp = sc * state[(size_t)row*H + t + j*TPB];
    float m = sp * g[j];
    mxv[j] = m; smm += m*m;
  }
  smm = rowsum(smm, sbuf);
  float mxn = fmaxf(sqrtf(smm), 1e-10f);
  float tt = tanhf(mxn / hn * atanh_c(hn));
  float sfac = tt / mxn;           // result = sfac * mx ; norm = |tt|
  float rn = fabsf(tt);
  float mb = 1.0f - 1e-4f;
  float psc = (rn > mb) ? (mb / (rn + 1e-10f)) : 1.0f;
  for (int j = 0; j < per; ++j)
    rp[(size_t)row*H + t + j*TPB] = (_Float16)(psc * sfac * mxv[j]);
  if (t == 0) rpnorm[row] = fmaxf(rn * psc, 1e-10f);
}

// ---------------- final combine ----------------

__global__ __launch_bounds__(TPB) void k_final(
    const float* __restrict__ Hh, const float* __restrict__ Hx,
    const float* __restrict__ bh, const float* __restrict__ state,
    const float* __restrict__ sscale, const float* __restrict__ snorm,
    const float* __restrict__ rpnorm, const float* __restrict__ xnorm,
    const _Float16* __restrict__ zbuf, int H, float* __restrict__ out)
{
  __shared__ float sbuf[TPB];
  const int row = blockIdx.x;
  const int t = threadIdx.x;
  const int per = H / TPB;
  const size_t base = (size_t)row * H;
  float u[GMAX], v[GMAX], bv[GMAX], spv[GMAX], zv[GMAX];
  const float sc = sscale[row];
  float s1 = 0.f, s2 = 0.f;
  for (int j = 0; j < per; ++j) {
    int i = t + j*TPB;
    float a = Hh[base + i], c = Hx[base + i];
    u[j] = a; v[j] = c;
    bv[j] = bh[i];
    spv[j] = sc * state[base + i];
    zv[j] = (float)zbuf[base + i];
    s1 += a*a; s2 += c*c;
  }
  s1 = rowsum(s1, sbuf);
  s2 = rowsum(s2, sbuf);
  const float rn = rpnorm[row], xn = xnorm[row], hn = snorm[row];
  // Th = mob_matmul(rph, Wh), Tx = mob_matmul(hyp_x, Uh)
  float nh = fmaxf(sqrtf(s1), 1e-10f);
  float th = tanhf(nh / rn * atanh_c(rn));
  float ah = th / nh;
  float nx = fmaxf(sqrtf(s2), 1e-10f);
  float tx = tanhf(nx / xn * atanh_c(xn));
  float ax = tx / nx;
  // a1 = mob_add(Th, Tx)
  float uv = 0.f;
  for (int j = 0; j < per; ++j) { u[j] *= ah; v[j] *= ax; uv += u[j]*v[j]; }
  uv = rowsum(uv, sbuf);
  float u2 = th*th, v2 = tx*tx;
  float den = 1.f + 2.f*uv + u2*v2 + 1e-10f;
  float cu = 1.f + 2.f*uv + v2, cv = 1.f - u2;
  float duv = 0.f, d2 = 0.f, db = 0.f;
  for (int j = 0; j < per; ++j) {
    float a1 = (cu*u[j] + cv*v[j]) / den;
    u[j] = a1;
    duv += a1*bv[j]; d2 += a1*a1; db += bv[j]*bv[j];
  }
  duv = rowsum(duv, sbuf);
  d2  = rowsum(d2,  sbuf);
  db  = rowsum(db,  sbuf);
  // a2 = mob_add(a1, bh) ; project(.., 1e-5) then project(.., 1e-4)
  float den2 = 1.f + 2.f*duv + d2*db + 1e-10f;
  float cu2 = 1.f + 2.f*duv + db, cv2 = 1.f - d2;
  float nn = 0.f;
  for (int j = 0; j < per; ++j) {
    float a2 = (cu2*u[j] + cv2*bv[j]) / den2;
    u[j] = a2; nn += a2*a2;
  }
  nn = rowsum(nn, sbuf);
  float n2 = sqrtf(nn);
  float mb5 = 1.0f - 1e-5f;
  float ps = (n2 > mb5) ? (mb5 / (n2 + 1e-10f)) : 1.0f;   // _one_transform proj
  float n2b = n2 * ps;
  float mb4 = 1.0f - 1e-4f;
  float ps2 = (n2b > mb4) ? (mb4 / (n2b + 1e-10f)) : 1.0f; // h_tilde proj 1e-4
  float htn = n2b * ps2;
  float hts = ps * ps2;
  for (int j = 0; j < per; ++j) u[j] *= hts;   // u = h_tilde
  // m = mob_add(-state_p, h_tilde)
  float uvm = 0.f;
  for (int j = 0; j < per; ++j) uvm += (-spv[j]) * u[j];
  uvm = rowsum(uvm, sbuf);
  float u2m = hn*hn, v2m = htn*htn;
  float denm = 1.f + 2.f*uvm + u2m*v2m + 1e-10f;
  float cum = 1.f + 2.f*uvm + v2m, cvm = 1.f - u2m;
  float svn = 0.f;
  for (int j = 0; j < per; ++j) {
    float m = (cum * (-spv[j]) + cvm * u[j]) / denm;
    v[j] = m; svn += m*m;
  }
  svn = rowsum(svn, sbuf);
  // upd = mob_pw(m, z)
  float smx = 0.f;
  for (int j = 0; j < per; ++j) { float m = v[j] * zv[j]; u[j] = m; smx += m*m; }
  smx = rowsum(smx, sbuf);
  float vn_  = fmaxf(sqrtf(svn), 1e-10f);
  float mxn_ = fmaxf(sqrtf(smx), 1e-10f);
  float tt = tanhf(mxn_ / vn_ * atanh_c(vn_));
  float ufac = tt / mxn_;
  // new_h = project(mob_add(state_p, upd), 1e-5)
  float uvf = 0.f;
  for (int j = 0; j < per; ++j) { u[j] *= ufac; uvf += spv[j] * u[j]; }
  uvf = rowsum(uvf, sbuf);
  float u2f = hn*hn, v2f = tt*tt;
  float denf = 1.f + 2.f*uvf + u2f*v2f + 1e-10f;
  float cuf = 1.f + 2.f*uvf + v2f, cvf = 1.f - u2f;
  float nres = 0.f;
  for (int j = 0; j < per; ++j) {
    float r = (cuf*spv[j] + cvf*u[j]) / denf;
    u[j] = r; nres += r*r;
  }
  nres = rowsum(nres, sbuf);
  float nr = sqrtf(nres);
  float psf = (nr > mb5) ? (mb5 / (nr + 1e-10f)) : 1.0f;
  for (int j = 0; j < per; ++j)
    out[base + t + j*TPB] = psf * u[j];
}

// ---------------- launch ----------------

extern "C" void kernel_launch(void* const* d_in, const int* in_sizes, int n_in,
                              void* d_out, int out_size, void* d_ws, size_t ws_size,
                              hipStream_t stream) {
  (void)n_in; (void)out_size; (void)ws_size;
  const float* inputs = (const float*)d_in[0];
  const float* state  = (const float*)d_in[1];
  const float* Wz = (const float*)d_in[2];
  const float* Uz = (const float*)d_in[3];
  const float* bz = (const float*)d_in[4];
  const float* Wr = (const float*)d_in[5];
  const float* Ur = (const float*)d_in[6];
  const float* br = (const float*)d_in[7];
  const float* Wh = (const float*)d_in[8];
  const float* Uh = (const float*)d_in[9];
  const float* bh = (const float*)d_in[10];
  float* out = (float*)d_out;

  const int H = in_sizes[4];          // bz has H elements
  const int D = in_sizes[3] / H;      // Uz is D x H
  const int B = in_sizes[0] / D;      // inputs is B x D

  char* ws = (char*)d_ws;
  size_t off = 0;
  auto walloc = [&](size_t bytes) -> void* {
    void* p = ws + off;
    off += (bytes + 255) & ~(size_t)255;
    return p;
  };
  _Float16* Xh   = (_Float16*)walloc((size_t)B * D * 2);
  _Float16* Sph  = (_Float16*)walloc((size_t)B * H * 2);
  _Float16* rp   = (_Float16*)walloc((size_t)B * H * 2);
  _Float16* zb   = (_Float16*)walloc((size_t)B * H * 2);
  float*    Gh   = (float*)walloc((size_t)B * H * 4);
  float*    Gx   = (float*)walloc((size_t)B * H * 4);
  _Float16* WtWz = (_Float16*)walloc((size_t)H * H * 2);
  _Float16* WtUz = (_Float16*)walloc((size_t)D * H * 2);
  _Float16* WtWr = (_Float16*)walloc((size_t)H * H * 2);
  _Float16* WtUr = (_Float16*)walloc((size_t)D * H * 2);
  _Float16* WtWh = (_Float16*)walloc((size_t)H * H * 2);
  _Float16* WtUh = (_Float16*)walloc((size_t)D * H * 2);
  float* xnorm  = (float*)walloc((size_t)B * 4);
  float* snorm  = (float*)walloc((size_t)B * 4);
  float* sscale = (float*)walloc((size_t)B * 4);
  float* rpnorm = (float*)walloc((size_t)B * 4);

  // 1) preprocess rows
  k_pre<<<B, TPB, 0, stream>>>(inputs, state, D, H, Xh, Sph, xnorm, snorm, sscale);

  // 2) weights -> f16, transposed to [N][K]
  dim3 tb(32, 8);
  k_wt<<<dim3(H/32, H/32), tb, 0, stream>>>(Wz, WtWz, H, H);
  k_wt<<<dim3(H/32, D/32), tb, 0, stream>>>(Uz, WtUz, D, H);
  k_wt<<<dim3(H/32, H/32), tb, 0, stream>>>(Wr, WtWr, H, H);
  k_wt<<<dim3(H/32, D/32), tb, 0, stream>>>(Ur, WtUr, D, H);
  k_wt<<<dim3(H/32, H/32), tb, 0, stream>>>(Wh, WtWh, H, H);
  k_wt<<<dim3(H/32, D/32), tb, 0, stream>>>(Uh, WtUh, D, H);

  dim3 gg(H / BN, B / BM);

  // 3) z gate
  k_gemm<<<gg, TPB, 0, stream>>>(Sph, WtWz, Gh, B, H, H);
  k_gemm<<<gg, TPB, 0, stream>>>(Xh,  WtUz, Gx, B, H, D);
  k_zgate<<<B, TPB, 0, stream>>>(Gh, Gx, bz, snorm, xnorm, H, zb);

  // 4) r gate + r_point_h
  k_gemm<<<gg, TPB, 0, stream>>>(Sph, WtWr, Gh, B, H, H);
  k_gemm<<<gg, TPB, 0, stream>>>(Xh,  WtUr, Gx, B, H, D);
  k_rgate<<<B, TPB, 0, stream>>>(Gh, Gx, br, state, sscale, snorm, xnorm, H,
                                 rp, rpnorm);

  // 5) h_tilde GEMMs + final combine
  k_gemm<<<gg, TPB, 0, stream>>>(rp, WtWh, Gh, B, H, H);
  k_gemm<<<gg, TPB, 0, stream>>>(Xh, WtUh, Gx, B, H, D);
  k_final<<<B, TPB, 0, stream>>>(Gh, Gx, bh, state, sscale, snorm, rpnorm,
                                 xnorm, zb, H, out);
}
